// PrototypeContrastiveLoss_75960791597443
// MI455X (gfx1250) — matrix operand order repaired and checked by reading, not compiled, not run on hardware
//
#include <hip/hip_runtime.h>
#include <math.h>

// ---------------------------------------------------------------------------
// PrototypeContrastiveLoss for MI455X (gfx1250, wave32, WMMA)
//
//   K1 prep_prompts   : z_prompts = l2norm(prompts)            -> f16 [Cp x D]
//   K2 prep_features  : enhanced_features rows (norm as needed)-> f16 [Nmax x D]
//   K3 class_weights  : class-balanced raw weights w[c]
//   K4 weight_scale   : wscale = C / sum(w); zero accumulators
//   K5 sim_loss       : A tile staged to LDS (async-to-LDS when available),
//                       WMMA GEMM 16x1024 tile, sim tile kept in LDS,
//                       top-10 hard-negative threshold + masked LSE + loss
//   K6 finalize       : out = loss_sum / max(n_valid, 1)
// ---------------------------------------------------------------------------

typedef __attribute__((ext_vector_type(16))) _Float16 v16h;
typedef __attribute__((ext_vector_type(8)))  _Float16 v8h;
typedef __attribute__((ext_vector_type(8)))  float    v8f;
typedef __attribute__((ext_vector_type(4)))  int      v4i;

#define D_DIM     1024
#define C_DIM     1000
#define CP        1024          // C padded for WMMA tiling
#define TEMP_INV  10.0f         // 1 / TEMPERATURE
#define HARDK     10
#define EPSV      1e-8f
#define LOG_BETA  (-0.0010005003335835335f)  // ln(0.999)

// Padded A-tile row stride in halfs: 1032 halfs = 2064 B = 516 dwords.
// 516 % 64 = 4 -> the 16 row-strided lanes of an A-fragment read hit
// banks 4*r .. 4*r+3 (all 64 banks), i.e. conflict-free ds_load_b128.
#define A_STRIDE_H 1032

#if defined(__has_builtin)
#  if __has_builtin(__builtin_amdgcn_global_load_async_to_lds_b128)
#    define HAVE_ASYNC_LDS 1
#  endif
#endif

__device__ __forceinline__ float waveMax(float v) {
#pragma unroll
    for (int off = 16; off; off >>= 1) v = fmaxf(v, __shfl_xor(v, off, 32));
    return v;
}
__device__ __forceinline__ float waveSum(float v) {
#pragma unroll
    for (int off = 16; off; off >>= 1) v += __shfl_xor(v, off, 32);
    return v;
}

// ---- K1: normalize prompts -> f16, zero-pad rows [C, CP) ------------------
__global__ void prep_prompts(const float* __restrict__ prompts,
                             _Float16* __restrict__ Pb) {
    const int c = blockIdx.x, t = threadIdx.x;
    __shared__ float red[256];
    _Float16* dst = Pb + (size_t)c * D_DIM;
    if (c >= C_DIM) {
        for (int k = t; k < D_DIM; k += 256) dst[k] = (_Float16)0.0f;
        return;
    }
    const float* src = prompts + (size_t)c * D_DIM;
    float ssq = 0.0f;
    for (int k = t; k < D_DIM; k += 256) { float x = src[k]; ssq += x * x; }
    red[t] = ssq; __syncthreads();
    for (int s = 128; s > 0; s >>= 1) { if (t < s) red[t] += red[t + s]; __syncthreads(); }
    const float inv = rsqrtf(red[0] + 1e-24f);
    for (int k = t; k < D_DIM; k += 256) dst[k] = (_Float16)(src[k] * inv);
}

// ---- K2: build enhanced_features rows -> f16, zero rows >= N --------------
__global__ void prep_features(const float* __restrict__ img,
                              const float* __restrict__ memf,
                              const int* __restrict__ mem_ptr_p,
                              int B, _Float16* __restrict__ Xb) {
    const int r = blockIdx.x, t = threadIdx.x;
    const int mp = *mem_ptr_p;            // uniform scalar load
    const int N  = B + mp + B;
    _Float16* dst = Xb + (size_t)r * D_DIM;
    if (r >= N) {
        for (int k = t; k < D_DIM; k += 256) dst[k] = (_Float16)0.0f;
        return;
    }
    const float* src; bool norm;
    if (r < B)            { src = img  + (size_t)r * D_DIM;            norm = true;  }
    else {
        int r2 = r - B;
        if (r2 < mp)      { src = memf + (size_t)r2 * D_DIM;           norm = false; }
        else              { src = img  + (size_t)(r2 - mp) * D_DIM;    norm = true;  }
    }
    float inv = 1.0f;
    __shared__ float red[256];
    if (norm) {  // uniform per block
        float ssq = 0.0f;
        for (int k = t; k < D_DIM; k += 256) { float x = src[k]; ssq += x * x; }
        red[t] = ssq; __syncthreads();
        for (int s = 128; s > 0; s >>= 1) { if (t < s) red[t] += red[t + s]; __syncthreads(); }
        inv = rsqrtf(red[0] + 1e-24f);
    }
    for (int k = t; k < D_DIM; k += 256) dst[k] = (_Float16)(src[k] * inv);
}

// ---- K3: per-class counts -> class-balanced raw weights -------------------
__global__ void class_weights(const int* __restrict__ labels, int B,
                              float* __restrict__ w) {
    const int c = blockIdx.x, t = threadIdx.x;
    __shared__ int red[256];
    int cnt = 0;
    for (int b = t; b < B; b += 256) cnt += labels[(size_t)b * C_DIM + c];
    red[t] = cnt; __syncthreads();
    for (int s = 128; s > 0; s >>= 1) { if (t < s) red[t] += red[t + s]; __syncthreads(); }
    if (t == 0) {
        float eff = 1.0f - __expf((float)red[0] * LOG_BETA);   // 1 - beta^count
        w[c] = 0.001f / (eff + EPSV);                          // (1-beta)/(...)
    }
}

// ---- K4: weight normalization scale + zero accumulators -------------------
__global__ void weight_scale(const float* __restrict__ w,
                             float* __restrict__ wscale,
                             float* __restrict__ loss_acc,
                             int* __restrict__ valid_acc) {
    const int t = threadIdx.x;
    __shared__ float red[256];
    float s = 0.0f;
    for (int c = t; c < C_DIM; c += 256) s += w[c];
    red[t] = s; __syncthreads();
    for (int k = 128; k > 0; k >>= 1) { if (t < k) red[t] += red[t + k]; __syncthreads(); }
    if (t == 0) { *wscale = (float)C_DIM / red[0]; *loss_acc = 0.0f; *valid_acc = 0; }
}

// ---- K5: fused WMMA GEMM tile + hard-neg top-10 + masked LSE + loss -------
__global__ __launch_bounds__(256) void sim_loss(
    const _Float16* __restrict__ Xb, const _Float16* __restrict__ Pb,
    const int* __restrict__ labels, const int* __restrict__ mlabels,
    const int* __restrict__ mem_ptr_p,
    const float* __restrict__ w, const float* __restrict__ wscale_p,
    float* __restrict__ loss_acc, int* __restrict__ valid_acc, int B) {

    // 64 KB LDS, time-multiplexed:
    //   phase 1: padded f16 A tile, 16 rows x A_STRIDE_H halfs (33 KB)
    //   phase 2: f32 sim tile, 16 rows x 1024 cols (64 KB)
    __shared__ __align__(16) unsigned char ldsRaw[16 * CP * 4];
    _Float16* aT   = (_Float16*)ldsRaw;
    float*    simF = (float*)ldsRaw;

    const int mp = *mem_ptr_p;
    const int N  = B + mp + B;
    const int r0 = blockIdx.x * 16;
    if (r0 >= N) return;

    const int t     = threadIdx.x;
    const int lane  = t & 31;
    const int wv    = t >> 5;                // wave 0..7
    const int mrow  = lane & 15;
    const int hi    = lane >> 4;             // lane half: 0 or 1
    const int nbase = wv * 128;              // this wave's 128 output columns

    // ---- stage A tile (16 rows x 1024 halfs) into padded LDS --------------
    // 2048 chunks of 16 B; 8 chunks per thread.
    for (int c = t; c < 16 * 128; c += 256) {
        const int row = c >> 7, col = c & 127;
        const v8h* g = (const v8h*)(Xb + (size_t)(r0 + row) * D_DIM) + col;
        _Float16*  l = aT + row * A_STRIDE_H + col * 8;
#ifdef HAVE_ASYNC_LDS
        __builtin_amdgcn_global_load_async_to_lds_b128(
            (v4i*)(uintptr_t)g, (v4i*)l, 0, 0);
#else
        *(v8h*)l = *g;
#endif
    }
#ifdef HAVE_ASYNC_LDS
#  if __has_builtin(__builtin_amdgcn_s_wait_asynccnt)
    __builtin_amdgcn_s_wait_asynccnt(0);
#  else
    asm volatile("s_wait_asynccnt 0" ::: "memory");
#  endif
#endif
    __syncthreads();

    // ---- WMMA K-loop: A from LDS, B from global (L2-resident prompts) -----
    // A fragment per-lane K chunk: {k0..k0+15}, k0 = 16*hi
    const _Float16* aRowL = aT + mrow * A_STRIDE_H + (hi << 4);
    const _Float16* bBase = Pb + (size_t)(nbase + mrow) * D_DIM + (hi << 4);

    v8f acc[8] = {};
    for (int kk = 0; kk < D_DIM; kk += 32) {
        v8h a0 = *(const v8h*)(aRowL + kk);          // ds_load_b128
        v8h a1 = *(const v8h*)(aRowL + kk + 8);      // ds_load_b128
        v16h afrag = __builtin_shufflevector(a0, a1,
            0,1,2,3,4,5,6,7,8,9,10,11,12,13,14,15);
        // Load all 8 B fragments first (one clause, partial waits),
        // then issue the 8 WMMAs so loads overlap matrix issue.
        v16h bf[8];
#pragma unroll
        for (int j = 0; j < 8; j++)
            bf[j] = *(const v16h*)(bBase + (size_t)(j * 16) * D_DIM + kk);
#pragma unroll
        for (int j = 0; j < 8; j++)
            acc[j] = __builtin_amdgcn_wmma_f32_16x16x32_f16(
                false, afrag, false, bf[j], (short)0, acc[j], false, false);
    }

    __syncthreads();   // all waves done reading aT before sim overwrites it

    // Scatter C/D fragments to LDS with 1/T scaling. m = i + 8*hi, n = lane&15.
#pragma unroll
    for (int j = 0; j < 8; j++) {
        const int n = nbase + j * 16 + mrow;
#pragma unroll
        for (int i = 0; i < 8; i++) {
            const int m = i + (hi << 3);
            simF[m * CP + n] = acc[j][i] * TEMP_INV;
        }
    }
    __syncthreads();

    const float wscale = *wscale_p;

    // One wave per row; 8 waves cover 16 rows in two passes.
    for (int lr = wv; lr < 16; lr += 8) {
        const int r = r0 + lr;
        if (r >= N) continue;

        const int* lrow;
        if (r < B) lrow = labels + (size_t)r * C_DIM;
        else {
            int r2 = r - B;
            lrow = (r2 < mp) ? (mlabels + (size_t)r2 * C_DIM)
                             : (labels + (size_t)(r2 - mp) * C_DIM);
        }
        const float* srow = simF + lr * CP;

        // positive-class bitmask over this lane's strided columns
        unsigned posbits = 0u;
        for (int j = 0; j < 32; j++) {
            const int c = lane + (j << 5);
            if (c >= C_DIM) break;
            if (lrow[c] > 0) posbits |= (1u << j);
        }

        // 10th-largest non-positive similarity via 10 wave-max passes
        float thr = __builtin_inff();
        for (int it = 0; it < HARDK; it++) {
            float m = -__builtin_inff();
            for (int j = 0; j < 32; j++) {
                const int c = lane + (j << 5);
                if (c >= C_DIM) break;
                if ((posbits >> j) & 1u) continue;
                const float s = srow[c];
                if (s < thr) m = fmaxf(m, s);
            }
            thr = waveMax(m);
        }

        // masked logsumexp over pos | (neg & sim >= thr)
        float rmax = -__builtin_inff();
        for (int j = 0; j < 32; j++) {
            const int c = lane + (j << 5);
            if (c >= C_DIM) break;
            const float s = srow[c];
            if (((posbits >> j) & 1u) || s >= thr) rmax = fmaxf(rmax, s);
        }
        rmax = waveMax(rmax);
        float se = 0.0f;
        for (int j = 0; j < 32; j++) {
            const int c = lane + (j << 5);
            if (c >= C_DIM) break;
            const float s = srow[c];
            if (((posbits >> j) & 1u) || s >= thr) se += __expf(s - rmax);
        }
        se = waveSum(se);
        const float lse = rmax + __logf(se);

        // weighted positive loss
        float lnum = 0.0f, wsum = 0.0f;
        for (int j = 0; j < 32; j++) {
            const int c = lane + (j << 5);
            if (c >= C_DIM) break;
            if (!((posbits >> j) & 1u)) continue;
            const float wc = w[c] * wscale;
            wsum += wc;
            lnum += wc * (lse - srow[c]);
        }
        lnum = waveSum(lnum);
        wsum = waveSum(wsum);
        if (lane == 0 && wsum > 0.0f) {
            atomicAdd(loss_acc, lnum / (wsum + EPSV));
            atomicAdd(valid_acc, 1);
        }
    }
}

// ---- K6: mean over valid rows ---------------------------------------------
__global__ void finalize(const float* __restrict__ loss_acc,
                         const int* __restrict__ valid_acc,
                         float* __restrict__ out) {
    int nv = *valid_acc;
    if (nv < 1) nv = 1;
    out[0] = *loss_acc / (float)nv;
}

// ---------------------------------------------------------------------------
extern "C" void kernel_launch(void* const* d_in, const int* in_sizes, int n_in,
                              void* d_out, int out_size, void* d_ws, size_t ws_size,
                              hipStream_t stream) {
    const float* img     = (const float*)d_in[0];   // [B, D]
    const float* prompts = (const float*)d_in[1];   // [C, D]
    const int*   labels  = (const int*)  d_in[2];   // [B, C]
    const float* memf    = (const float*)d_in[3];   // [M, D]
    const int*   meml    = (const int*)  d_in[4];   // [M, C]
    const int*   mpp     = (const int*)  d_in[5];   // scalar mem_ptr (device)

    const int B    = in_sizes[0] / D_DIM;
    const int M    = in_sizes[3] / D_DIM;
    const int Nmax = B + M;                 // worst-case enhanced rows

    // Workspace layout (16B-aligned chunks)
    char*     ws  = (char*)d_ws;
    _Float16* Xb  = (_Float16*)ws;                       // Nmax*D f16
    size_t    off = (size_t)Nmax * D_DIM * sizeof(_Float16);
    _Float16* Pb  = (_Float16*)(ws + off);               // CP*D f16
    off += (size_t)CP * D_DIM * sizeof(_Float16);
    float* wcls      = (float*)(ws + off);  off += CP * sizeof(float);
    float* wscale    = (float*)(ws + off);  off += 16;
    float* loss_acc  = (float*)(ws + off);  off += 16;
    int*   valid_acc = (int*)  (ws + off);  off += 16;

    prep_prompts <<<CP,   256, 0, stream>>>(prompts, Pb);
    prep_features<<<Nmax, 256, 0, stream>>>(img, memf, mpp, B, Xb);
    class_weights<<<C_DIM,256, 0, stream>>>(labels, B, wcls);
    weight_scale <<<1,    256, 0, stream>>>(wcls, wscale, loss_acc, valid_acc);
    sim_loss     <<<(Nmax + 15) / 16, 256, 0, stream>>>(
        Xb, Pb, labels, meml, mpp, wcls, wscale, loss_acc, valid_acc, B);
    finalize     <<<1, 1, 0, stream>>>(loss_acc, valid_acc, (float*)d_out);
}
